// LogisticRegressionPerStock_1700807050061
// MI455X (gfx1250) — matrix operand order repaired and checked by reading, not compiled, not run on hardware
//
#include <hip/hip_runtime.h>
#include <stdint.h>

// ---------------------------------------------------------------------------
// LogisticRegressionPerStock for MI455X (gfx1250, wave32, WMMA)
//   Stage 1: h[b,s] = sigmoid(sum_t x[b,t,s] * W1[s,t] + b1[s])   (mem-bound)
//   Stage 2: out    = h @ W2^T + b2  -- 1024x8192x8192 NT GEMM via
//            split-precision bf16x3 V_WMMA_F32_16X16X32_BF16
// ---------------------------------------------------------------------------

typedef __attribute__((ext_vector_type(16))) __bf16       v16bf;
typedef __attribute__((ext_vector_type(8)))  float        v8f;
typedef __attribute__((ext_vector_type(4)))  float        f32x4;
typedef __attribute__((ext_vector_type(4)))  unsigned int u32x4;
typedef __attribute__((ext_vector_type(2)))  unsigned int u32x2;

#define S_DIM 8192
#define T_DIM 24
#define B_DIM 1024

#define BM  128
#define BN  128
#define BK  32
#define BKP 40   // padded LDS stride in halves (40*2B = 80B -> conflict-free)

// ---------------------------------------------------------------------------
// Stage 1: per-stock logistic heads. Coalesced over s; W1 chunk staged in LDS.
// ---------------------------------------------------------------------------
__global__ __launch_bounds__(256)
void stock_head_kernel(const float* __restrict__ x,   // [B, T, S]
                       const float* __restrict__ W1,  // [S, T]
                       const float* __restrict__ b1,  // [S]
                       float* __restrict__ h) {       // [B, S]
  __shared__ float w1s[256][T_DIM + 1];  // +1 pad: conflict-free row reads
  const int tid = threadIdx.x;
  const int s0  = blockIdx.x * 256;
  const int b   = blockIdx.y;

  for (int i = tid; i < 256 * T_DIM; i += 256) {
    int r = i / T_DIM, c = i % T_DIM;
    w1s[r][c] = W1[(size_t)(s0 + r) * T_DIM + c];
  }
  __syncthreads();

  const int s = s0 + tid;
  float acc = b1[s];
  const float* xp = x + (size_t)b * T_DIM * S_DIM + s;
#pragma unroll
  for (int t = 0; t < T_DIM; ++t)
    acc = fmaf(xp[(size_t)t * S_DIM], w1s[tid][t], acc);

  h[(size_t)b * S_DIM + s] = 1.0f / (1.0f + __expf(-acc));
}

// ---------------------------------------------------------------------------
// Stage 2 helpers
// ---------------------------------------------------------------------------
__device__ __forceinline__ void split_bf16(float v, uint16_t& hi, uint16_t& lo) {
  unsigned u = __float_as_uint(v);
  hi = (uint16_t)(u >> 16);                              // truncate to bf16
  float r = v - __uint_as_float(u & 0xFFFF0000u);        // exact residual
  lo = (uint16_t)(__float_as_uint(r) >> 16);             // residual ~2^-16 rel
}

// A fragment (16x32 bf16): lane L row = l16; lanes<16 hold K={0..7,16..23},
// lanes>=16 hold K={8..15,24..31}  -> two 16B runs at K = h16*8 and h16*8+16.
__device__ __forceinline__ v16bf load_frag_a(const uint16_t* base, int row, int h16) {
  int off = row * BKP + h16 * 8;
  union { v16bf v; u32x4 q[2]; } u;
  u.q[0] = *(const u32x4*)(base + off);
  u.q[1] = *(const u32x4*)(base + off + 16);
  return u.v;
}

// B fragment (32x16 bf16): lane L col = l16; lanes<16 hold K={0..7,16..23},
// lanes>=16 hold K={8..15,24..31} -> two 16B runs at K = h16*8 and h16*8+16.
__device__ __forceinline__ v16bf load_frag_b(const uint16_t* base, int col, int h16) {
  int off = col * BKP + h16 * 8;
  union { v16bf v; u32x4 q[2]; } u;
  u.q[0] = *(const u32x4*)(base + off);
  u.q[1] = *(const u32x4*)(base + off + 16);
  return u.v;
}

// ---------------------------------------------------------------------------
// Stage 2: out[m,n] = sum_k h[m,k] * W2[n,k] + b2[n]
// Block: 128x128 C-tile, 8 waves in 2(M)x4(N); wave tile = 64x32 = 4x2 WMMA tiles.
// ---------------------------------------------------------------------------
__global__ __launch_bounds__(256)
void gemm_bf16x3_kernel(const float* __restrict__ h,    // [1024, 8192]
                        const float* __restrict__ W2,   // [8192, 8192]
                        const float* __restrict__ b2,   // [8192]
                        float* __restrict__ out) {      // [1024, 8192]
  __shared__ uint16_t aHi[BM * BKP], aLo[BM * BKP];
  __shared__ uint16_t bHi[BN * BKP], bLo[BN * BKP];

  const int tid  = threadIdx.x;
  const int lane = tid & 31;
  const int l16  = lane & 15;
  const int h16  = lane >> 4;
  const int wave = tid >> 5;

  const int bm0 = blockIdx.y * BM;
  const int bn0 = blockIdx.x * BN;

  const int mBase = (wave >> 2) * 64;  // wave M offset within block tile
  const int nBase = (wave & 3) * 32;   // wave N offset within block tile

  v8f acc[4][2];
#pragma unroll
  for (int i = 0; i < 4; ++i)
#pragma unroll
    for (int j = 0; j < 2; ++j) acc[i][j] = (v8f)0.0f;

  // Global-load geometry: 128x32 f32 tile = 1024 f32x4; 4 per thread.
  const int rowB0 = tid >> 3;        // 0..31, + 32*it
  const int kOff  = (tid & 7) * 4;   // 0..28

  f32x4 ra[4], rb[4];
  // prologue: stage 0
#pragma unroll
  for (int it = 0; it < 4; ++it) {
    int r = rowB0 + 32 * it;
    ra[it] = *(const f32x4*)(h  + (size_t)(bm0 + r) * S_DIM + kOff);
    rb[it] = *(const f32x4*)(W2 + (size_t)(bn0 + r) * S_DIM + kOff);
  }

  const int nStages = S_DIM / BK;  // 256
  for (int s = 0; s < nStages; ++s) {
    __syncthreads();  // previous stage's LDS reads complete

    // split f32 -> (hi, lo) bf16 and store to LDS
#pragma unroll
    for (int it = 0; it < 4; ++it) {
      int r = rowB0 + 32 * it;
      uint16_t h4[4], l4[4], h4b[4], l4b[4];
#pragma unroll
      for (int c = 0; c < 4; ++c) {
        split_bf16(ra[it][c], h4[c], l4[c]);
        split_bf16(rb[it][c], h4b[c], l4b[c]);
      }
      u32x2 ph  = { (unsigned)h4[0]  | ((unsigned)h4[1]  << 16),
                    (unsigned)h4[2]  | ((unsigned)h4[3]  << 16) };
      u32x2 pl  = { (unsigned)l4[0]  | ((unsigned)l4[1]  << 16),
                    (unsigned)l4[2]  | ((unsigned)l4[3]  << 16) };
      u32x2 phb = { (unsigned)h4b[0] | ((unsigned)h4b[1] << 16),
                    (unsigned)h4b[2] | ((unsigned)h4b[3] << 16) };
      u32x2 plb = { (unsigned)l4b[0] | ((unsigned)l4b[1] << 16),
                    (unsigned)l4b[2] | ((unsigned)l4b[3] << 16) };
      *(u32x2*)(aHi + r * BKP + kOff) = ph;
      *(u32x2*)(aLo + r * BKP + kOff) = pl;
      *(u32x2*)(bHi + r * BKP + kOff) = phb;
      *(u32x2*)(bLo + r * BKP + kOff) = plb;
    }
    __syncthreads();  // LDS tile visible

    // prefetch next stage's globals (overlaps with WMMA burst below)
    if (s + 1 < nStages) {
      int k0 = (s + 1) * BK;
#pragma unroll
      for (int it = 0; it < 4; ++it) {
        int r = rowB0 + 32 * it;
        ra[it] = *(const f32x4*)(h  + (size_t)(bm0 + r) * S_DIM + k0 + kOff);
        rb[it] = *(const f32x4*)(W2 + (size_t)(bn0 + r) * S_DIM + k0 + kOff);
      }
    }

    // bf16x3: acc += Ahi*Bhi + Ahi*Blo + Alo*Bhi   (Alo*Blo ~2^-32, dropped)
#pragma unroll
    for (int j = 0; j < 2; ++j) {
      v16bf bh = load_frag_b(bHi, nBase + j * 16 + l16, h16);
      v16bf bl = load_frag_b(bLo, nBase + j * 16 + l16, h16);
#pragma unroll
      for (int i = 0; i < 4; ++i) {
        v16bf ah = load_frag_a(aHi, mBase + i * 16 + l16, h16);
        v16bf al = load_frag_a(aLo, mBase + i * 16 + l16, h16);
        acc[i][j] = __builtin_amdgcn_wmma_f32_16x16x32_bf16(
            false, ah, false, bh, (short)0, acc[i][j], false, false);
        acc[i][j] = __builtin_amdgcn_wmma_f32_16x16x32_bf16(
            false, ah, false, bl, (short)0, acc[i][j], false, false);
        acc[i][j] = __builtin_amdgcn_wmma_f32_16x16x32_bf16(
            false, al, false, bh, (short)0, acc[i][j], false, false);
      }
    }
  }

  // Epilogue: C/D layout -> M = r + 8*h16 (VGPR r), N = l16; add b2, store.
#pragma unroll
  for (int j = 0; j < 2; ++j) {
    int gn = bn0 + nBase + j * 16 + l16;
    float bias = b2[gn];
#pragma unroll
    for (int i = 0; i < 4; ++i) {
      int gm = bm0 + mBase + i * 16 + h16 * 8;
#pragma unroll
      for (int r = 0; r < 8; ++r)
        out[(size_t)(gm + r) * S_DIM + gn] = acc[i][j][r] + bias;
    }
  }
}

// ---------------------------------------------------------------------------
extern "C" void kernel_launch(void* const* d_in, const int* in_sizes, int n_in,
                              void* d_out, int out_size, void* d_ws, size_t ws_size,
                              hipStream_t stream) {
  const float* x  = (const float*)d_in[0];  // [B, T, S]
  const float* W1 = (const float*)d_in[1];  // [S, T]
  const float* b1 = (const float*)d_in[2];  // [S]
  const float* W2 = (const float*)d_in[3];  // [S, S]
  const float* b2 = (const float*)d_in[4];  // [S]
  float* out = (float*)d_out;
  float* hbuf = (float*)d_ws;               // [B, S] fp32 = 32 MB scratch

  dim3 g1(S_DIM / 256, B_DIM);
  stock_head_kernel<<<g1, 256, 0, stream>>>(x, W1, b1, hbuf);

  dim3 g2(S_DIM / BN, B_DIM / BM);          // (64, 8)
  gemm_bf16x3_kernel<<<g2, 256, 0, stream>>>(hbuf, W2, b2, out);
}
